// Seq2Seq_44916767981986
// MI455X (gfx1250) — matrix operand order repaired and checked by reading, not compiled, not run on hardware
//
#include <hip/hip_runtime.h>

// ---------------------------------------------------------------------------
// Types
// ---------------------------------------------------------------------------
typedef __bf16 bf16_t;
typedef __attribute__((ext_vector_type(16))) __bf16 v16bf;
typedef __attribute__((ext_vector_type(8)))  __bf16 v8bf;
typedef __attribute__((ext_vector_type(8)))  float  v8f;
typedef __attribute__((ext_vector_type(2)))  float  v2f;

#define SLEN  48
#define TLEN  32
#define BATCH 32
#define EDIM  512
#define HDIM  1024
#define G4H   4096           // 4*H
#define VTGT  32000
#define DECM  ((TLEN - 1) * BATCH)   // 992

__device__ __forceinline__ float sigmoidf_(float x) { return 1.f / (1.f + __expf(-x)); }

// ---------------------------------------------------------------------------
// WMMA fragment loaders (layouts per CDNA5 ISA §7.12.2)
// A 16x32 bf16: lane L -> row M=L%16; elems 0..7 = K (L/16)*8+0..7,
//                                     elems 8..15 = 16+(L/16)*8+0..7
// B 32x16 bf16: lane L -> col N=L%16; elems 0..7 = K (L/16)*8+0..7,
//                                     elems 8..15 = 16+(L/16)*8+0..7
// ---------------------------------------------------------------------------
__device__ __forceinline__ v16bf load_a_frag(const bf16_t* __restrict__ p, int ld,
                                             int row, int k0, int lh) {
  const bf16_t* q = p + (size_t)row * ld + k0 + lh * 8;
  v8bf lo = *(const v8bf*)q;
  v8bf hi = *(const v8bf*)(q + 16);
  v16bf r;
#pragma unroll
  for (int i = 0; i < 8; ++i) { r[i] = lo[i]; r[i + 8] = hi[i]; }
  return r;
}

__device__ __forceinline__ v16bf load_b_frag(const bf16_t* __restrict__ p, int ld,
                                             int row, int k0, int lh) {
  const bf16_t* q = p + (size_t)row * ld + k0 + lh * 8;
  v8bf lo = *(const v8bf*)q;
  v8bf hi = *(const v8bf*)(q + 16);
  v16bf r;
#pragma unroll
  for (int i = 0; i < 8; ++i) { r[i] = lo[i]; r[i + 8] = hi[i]; }
  return r;
}

__device__ __forceinline__ v8f wmma_bf16(v16bf a, v16bf b, v8f c) {
  return __builtin_amdgcn_wmma_f32_16x16x32_bf16(false, a, false, b, (short)0, c,
                                                 false, false);
}

// ---------------------------------------------------------------------------
// Elementwise helpers
// ---------------------------------------------------------------------------
__global__ void k_zero_u32(unsigned int* __restrict__ p, int n) {
  int i = blockIdx.x * blockDim.x + threadIdx.x;
  if (i < n) p[i] = 0u;
}

__global__ void k_f32_to_bf16(const float* __restrict__ in, bf16_t* __restrict__ out, int n) {
  int i = blockIdx.x * blockDim.x + threadIdx.x;
  if (i < n) out[i] = (bf16_t)in[i];
}

// gather rows of an fp32 table by int32 index, convert to bf16
__global__ void k_gather_bf16(const float* __restrict__ table, const int* __restrict__ idx,
                              bf16_t* __restrict__ out, int nrows, int ncols) {
  int i = blockIdx.x * blockDim.x + threadIdx.x;
  if (i >= nrows * ncols) return;
  int r = i / ncols;
  int c = i - r * ncols;
  out[i] = (bf16_t)table[(size_t)idx[r] * ncols + c];
}

// ---------------------------------------------------------------------------
// Generic bf16 GEMM: out[M,N] (f32) = A[M,K](bf16) @ W[N,K](bf16)^T
// grid.x = N/16, grid.y = M/32, block = 32 (one wave -> 32x16 tile)
// ---------------------------------------------------------------------------
__global__ void __launch_bounds__(32)
k_gemm_bf16f32(const bf16_t* __restrict__ A, const bf16_t* __restrict__ W,
               float* __restrict__ out, int N, int K) {
  const int lane = threadIdx.x & 31;
  const int lh = lane >> 4, lm = lane & 15;
  const int n0 = blockIdx.x * 16;
  const int m0 = blockIdx.y * 32;
  v8f acc0 = {}, acc1 = {};
  for (int k0 = 0; k0 < K; k0 += 32) {
    v16bf a0 = load_a_frag(A, K, m0 + lm, k0, lh);
    v16bf a1 = load_a_frag(A, K, m0 + 16 + lm, k0, lh);
    v16bf bw = load_b_frag(W, K, n0 + lm, k0, lh);
    acc0 = wmma_bf16(a0, bw, acc0);
    acc1 = wmma_bf16(a1, bw, acc1);
  }
  const int n = n0 + lm;
#pragma unroll
  for (int v = 0; v < 8; ++v) {
    out[(size_t)(m0 + v + 8 * lh) * N + n]      = acc0[v];
    out[(size_t)(m0 + 16 + v + 8 * lh) * N + n] = acc1[v];
  }
}

// ---------------------------------------------------------------------------
// Fused LSTM cell body (batch = 32), one wave per 16 hidden columns.
// gates = [optional x@Wih^T] + hprev@Whh^T + [optional precomputed gx] + bias
// then  c' = sig(f)*c + sig(i)*tanh(g);  h' = sig(o)*tanh(c')
// Computes all 4 gate tiles (i,f,g,o) so the nonlinear update is local.
// ---------------------------------------------------------------------------
__device__ __forceinline__ void
lstm_cell_body(int n0, int lane,
               const bf16_t* __restrict__ x, const bf16_t* __restrict__ Wih, int Kx,
               const bf16_t* __restrict__ hprev, const bf16_t* __restrict__ Whh,
               const float* __restrict__ gx, const float* __restrict__ bias,
               float* __restrict__ c, bf16_t* __restrict__ hout) {
  const int lh = lane >> 4, lm = lane & 15;
  v8f acc[4][2] = {};

  if (x != nullptr) {
    for (int k0 = 0; k0 < Kx; k0 += 32) {
      v16bf a0 = load_a_frag(x, Kx, lm, k0, lh);
      v16bf a1 = load_a_frag(x, Kx, 16 + lm, k0, lh);
#pragma unroll
      for (int g = 0; g < 4; ++g) {
        v16bf bw = load_b_frag(Wih, Kx, g * HDIM + n0 + lm, k0, lh);
        acc[g][0] = wmma_bf16(a0, bw, acc[g][0]);
        acc[g][1] = wmma_bf16(a1, bw, acc[g][1]);
      }
    }
  }
  for (int k0 = 0; k0 < HDIM; k0 += 32) {
    v16bf a0 = load_a_frag(hprev, HDIM, lm, k0, lh);
    v16bf a1 = load_a_frag(hprev, HDIM, 16 + lm, k0, lh);
#pragma unroll
    for (int g = 0; g < 4; ++g) {
      v16bf bw = load_b_frag(Whh, HDIM, g * HDIM + n0 + lm, k0, lh);
      acc[g][0] = wmma_bf16(a0, bw, acc[g][0]);
      acc[g][1] = wmma_bf16(a1, bw, acc[g][1]);
    }
  }

  const int n = n0 + lm;
#pragma unroll
  for (int t = 0; t < 2; ++t) {
#pragma unroll
    for (int v = 0; v < 8; ++v) {
      const int m = t * 16 + v + 8 * lh;
      float gi = acc[0][t][v] + bias[n];
      float gf = acc[1][t][v] + bias[HDIM + n];
      float gg = acc[2][t][v] + bias[2 * HDIM + n];
      float go = acc[3][t][v] + bias[3 * HDIM + n];
      if (gx != nullptr) {
        const float* gxr = gx + (size_t)m * G4H;
        gi += gxr[n];
        gf += gxr[HDIM + n];
        gg += gxr[2 * HDIM + n];
        go += gxr[3 * HDIM + n];
      }
      float cold = c[(size_t)m * HDIM + n];
      float cn = sigmoidf_(gf) * cold + sigmoidf_(gi) * tanhf(gg);
      float hn = sigmoidf_(go) * tanhf(cn);
      c[(size_t)m * HDIM + n] = cn;
      hout[(size_t)m * HDIM + n] = (bf16_t)hn;
    }
  }
}

// ---------------------------------------------------------------------------
// Software-pipelined encoder step: one launch per pipeline tick t in [0, 48].
//   blockIdx.y == 0 : layer-0 cell at step t      (valid for t < 48)
//   blockIdx.y == 1 : layer-1 cell at step t - 1  (valid for t >= 1)
// These are independent: layer1(t-1) consumes hbuf0[t] written last tick,
// layer0(t) only reads hbuf0[t]. Halves the number of serialized launches
// on the recurrence critical path (49 vs 96) and doubles waves per launch.
// Branches are workgroup-uniform -> EXEC all ones for WMMA.
// grid = (HDIM/16, 2), block = 32.
// ---------------------------------------------------------------------------
__global__ void __launch_bounds__(32)
k_enc_pipe(int t,
           const float* __restrict__ gx0, const float* __restrict__ b0,
           const bf16_t* __restrict__ Whh0, bf16_t* __restrict__ hbuf0,
           float* __restrict__ c0,
           const bf16_t* __restrict__ Wih1, const bf16_t* __restrict__ Whh1,
           const float* __restrict__ b1, bf16_t* __restrict__ hbuf1,
           float* __restrict__ c1) {
  const int lane = threadIdx.x & 31;
  const int n0 = blockIdx.x * 16;
  if (blockIdx.y == 0) {
    if (t < SLEN) {
      lstm_cell_body(n0, lane, nullptr, nullptr, 0,
                     hbuf0 + (size_t)t * BATCH * HDIM, Whh0,
                     gx0 + (size_t)t * BATCH * G4H, b0,
                     c0, hbuf0 + (size_t)(t + 1) * BATCH * HDIM);
    }
  } else {
    if (t >= 1) {
      const int s = t - 1;  // layer-1 time step
      lstm_cell_body(n0, lane,
                     hbuf0 + (size_t)(s + 1) * BATCH * HDIM, Wih1, HDIM,
                     hbuf1 + (size_t)s * BATCH * HDIM, Whh1,
                     nullptr, b1,
                     c1, hbuf1 + (size_t)(s + 1) * BATCH * HDIM);
    }
  }
}

// ---------------------------------------------------------------------------
// Decoder cell: fully batched & elementwise.
// gates[m] = gx[m] + hg[m%32] + b ; c0 = c_enc[m%32]
// ---------------------------------------------------------------------------
__global__ void k_dec_cell(const float* __restrict__ gx, const float* __restrict__ hg,
                           const float* __restrict__ b, const float* __restrict__ c_enc,
                           bf16_t* __restrict__ hout_bf, float* __restrict__ hout_f, int M) {
  int i = blockIdx.x * blockDim.x + threadIdx.x;
  if (i >= M * HDIM) return;
  int m = i >> 10;            // /HDIM
  int n = i & (HDIM - 1);
  int mm = m & (BATCH - 1);   // tiled encoder state row
  const float* gxr = gx + (size_t)m * G4H;
  const float* hgr = hg + (size_t)mm * G4H;
  float gi = gxr[n] + hgr[n] + b[n];
  float gf = gxr[HDIM + n] + hgr[HDIM + n] + b[HDIM + n];
  float gg = gxr[2 * HDIM + n] + hgr[2 * HDIM + n] + b[2 * HDIM + n];
  float go = gxr[3 * HDIM + n] + hgr[3 * HDIM + n] + b[3 * HDIM + n];
  float cn = sigmoidf_(gf) * c_enc[(size_t)mm * HDIM + n] + sigmoidf_(gi) * tanhf(gg);
  float hn = sigmoidf_(go) * tanhf(cn);
  if (hout_bf) hout_bf[i] = (bf16_t)hn;
  if (hout_f)  hout_f[i]  = hn;
}

// ---------------------------------------------------------------------------
// FC: logits[992, 32000] = h[992,1024](f32) @ fcW[32000,1024](f32)^T + fcb
// Memory-bound on fcW (131MB fp32) -> read it once, use V_WMMA_F32_16X16X4_F32.
// grid.x = V/16 = 2000, grid.y = M/32 = 31, block = 32.
// ---------------------------------------------------------------------------
__global__ void __launch_bounds__(32)
k_fc_f32(const float* __restrict__ hf, const float* __restrict__ fcW,
         const float* __restrict__ fcb, float* __restrict__ out) {
  const int lane = threadIdx.x & 31;
  const int lh = lane >> 4, lm = lane & 15;
  const int n0 = blockIdx.x * 16;
  const int m0 = blockIdx.y * 32;
  v8f acc0 = {}, acc1 = {};
  const float* arow0 = hf + (size_t)(m0 + lm) * HDIM + 2 * lh;
  const float* arow1 = arow0 + (size_t)16 * HDIM;
  const float* brow  = fcW + (size_t)(n0 + lm) * HDIM + 2 * lh;
#pragma unroll 8
  for (int k0 = 0; k0 < HDIM; k0 += 4) {
    v2f a0 = *(const v2f*)(arow0 + k0);
    v2f a1 = *(const v2f*)(arow1 + k0);
    v2f bw = *(const v2f*)(brow + k0);
    acc0 = __builtin_amdgcn_wmma_f32_16x16x4_f32(false, a0, false, bw, (short)0, acc0,
                                                 false, false);
    acc1 = __builtin_amdgcn_wmma_f32_16x16x4_f32(false, a1, false, bw, (short)0, acc1,
                                                 false, false);
  }
  const int n = n0 + lm;
  const float bb = fcb[n];
#pragma unroll
  for (int v = 0; v < 8; ++v) {
    out[(size_t)(m0 + v + 8 * lh) * VTGT + n]      = acc0[v] + bb;
    out[(size_t)(m0 + 16 + v + 8 * lh) * VTGT + n] = acc1[v] + bb;
  }
}

// ---------------------------------------------------------------------------
// Host orchestration
// ---------------------------------------------------------------------------
static inline size_t alignup(size_t x) { return (x + 255) & ~(size_t)255; }

extern "C" void kernel_launch(void* const* d_in, const int* in_sizes, int n_in,
                              void* d_out, int out_size, void* d_ws, size_t ws_size,
                              hipStream_t stream) {
  (void)in_sizes; (void)n_in; (void)out_size; (void)ws_size;

  // Inputs (setup_inputs dict order)
  const int*   source   = (const int*)d_in[0];    // [48,32]
  const int*   target   = (const int*)d_in[1];    // [32,32]
  const float* e_emb    = (const float*)d_in[2];
  const float* e_Wih0   = (const float*)d_in[3];
  const float* e_Whh0   = (const float*)d_in[4];
  const float* e_b0     = (const float*)d_in[5];
  const float* e_Wih1   = (const float*)d_in[6];
  const float* e_Whh1   = (const float*)d_in[7];
  const float* e_b1     = (const float*)d_in[8];
  const float* d_emb    = (const float*)d_in[9];
  const float* fc_W     = (const float*)d_in[10];
  const float* fc_b     = (const float*)d_in[11];
  const float* dd_Wih0  = (const float*)d_in[12];
  const float* dd_Whh0  = (const float*)d_in[13];
  const float* dd_b0    = (const float*)d_in[14];
  const float* dd_Wih1  = (const float*)d_in[15];
  const float* dd_Whh1  = (const float*)d_in[16];
  const float* dd_b1    = (const float*)d_in[17];

  float* out = (float*)d_out;

  // Workspace bump allocator
  char* wsb = (char*)d_ws;
  size_t off = 0;
  auto alloc = [&](size_t bytes) -> void* {
    void* p = wsb + off;
    off = alignup(off + bytes);
    return p;
  };

  bf16_t* eWih0b = (bf16_t*)alloc((size_t)G4H * EDIM * 2);
  bf16_t* eWhh0b = (bf16_t*)alloc((size_t)G4H * HDIM * 2);
  bf16_t* eWih1b = (bf16_t*)alloc((size_t)G4H * HDIM * 2);
  bf16_t* eWhh1b = (bf16_t*)alloc((size_t)G4H * HDIM * 2);
  bf16_t* dWih0b = (bf16_t*)alloc((size_t)G4H * EDIM * 2);
  bf16_t* dWhh0b = (bf16_t*)alloc((size_t)G4H * HDIM * 2);
  bf16_t* dWih1b = (bf16_t*)alloc((size_t)G4H * HDIM * 2);
  bf16_t* dWhh1b = (bf16_t*)alloc((size_t)G4H * HDIM * 2);
  bf16_t* xs_bf  = (bf16_t*)alloc((size_t)SLEN * BATCH * EDIM * 2);
  float*  gx0    = (float*)alloc((size_t)SLEN * BATCH * G4H * 4);   // enc L0 x@Wih^T, all steps
  bf16_t* hbuf0  = (bf16_t*)alloc((size_t)(SLEN + 1) * BATCH * HDIM * 2);
  bf16_t* hbuf1  = (bf16_t*)alloc((size_t)(SLEN + 1) * BATCH * HDIM * 2);
  float*  c0     = (float*)alloc((size_t)BATCH * HDIM * 4);
  float*  c1     = (float*)alloc((size_t)BATCH * HDIM * 4);
  bf16_t* demb   = (bf16_t*)alloc((size_t)DECM * EDIM * 2);
  float*  gxd0   = (float*)alloc((size_t)DECM * G4H * 4);
  float*  gxd1   = (float*)alloc((size_t)DECM * G4H * 4);
  float*  hg0    = (float*)alloc((size_t)BATCH * G4H * 4);
  float*  hg1    = (float*)alloc((size_t)BATCH * G4H * 4);
  bf16_t* hd0b   = (bf16_t*)alloc((size_t)DECM * HDIM * 2);
  float*  hd1f   = (float*)alloc((size_t)DECM * HDIM * 4);

  const int ZT = 256;
  auto zgrid = [](int n) { return (n + 255) / 256; };

  // --- zero init: h0/c0 state slots + first output time-slice ---
  k_zero_u32<<<zgrid(BATCH * HDIM / 2), ZT, 0, stream>>>((unsigned*)hbuf0, BATCH * HDIM / 2);
  k_zero_u32<<<zgrid(BATCH * HDIM / 2), ZT, 0, stream>>>((unsigned*)hbuf1, BATCH * HDIM / 2);
  k_zero_u32<<<zgrid(BATCH * HDIM), ZT, 0, stream>>>((unsigned*)c0, BATCH * HDIM);
  k_zero_u32<<<zgrid(BATCH * HDIM), ZT, 0, stream>>>((unsigned*)c1, BATCH * HDIM);
  k_zero_u32<<<zgrid(BATCH * VTGT), ZT, 0, stream>>>((unsigned*)out, BATCH * VTGT);

  // --- convert recurrent weights to bf16 (amortized over many reads) ---
  k_f32_to_bf16<<<zgrid(G4H * EDIM), ZT, 0, stream>>>(e_Wih0, eWih0b, G4H * EDIM);
  k_f32_to_bf16<<<zgrid(G4H * HDIM), ZT, 0, stream>>>(e_Whh0, eWhh0b, G4H * HDIM);
  k_f32_to_bf16<<<zgrid(G4H * HDIM), ZT, 0, stream>>>(e_Wih1, eWih1b, G4H * HDIM);
  k_f32_to_bf16<<<zgrid(G4H * HDIM), ZT, 0, stream>>>(e_Whh1, eWhh1b, G4H * HDIM);
  k_f32_to_bf16<<<zgrid(G4H * EDIM), ZT, 0, stream>>>(dd_Wih0, dWih0b, G4H * EDIM);
  k_f32_to_bf16<<<zgrid(G4H * HDIM), ZT, 0, stream>>>(dd_Whh0, dWhh0b, G4H * HDIM);
  k_f32_to_bf16<<<zgrid(G4H * HDIM), ZT, 0, stream>>>(dd_Wih1, dWih1b, G4H * HDIM);
  k_f32_to_bf16<<<zgrid(G4H * HDIM), ZT, 0, stream>>>(dd_Whh1, dWhh1b, G4H * HDIM);

  // --- embedding gathers (bf16) ---
  k_gather_bf16<<<zgrid(SLEN * BATCH * EDIM), ZT, 0, stream>>>(e_emb, source, xs_bf,
                                                              SLEN * BATCH, EDIM);
  k_gather_bf16<<<zgrid(DECM * EDIM), ZT, 0, stream>>>(d_emb, target, demb, DECM, EDIM);

  // --- encoder layer-0 input GEMM batched over all 48 steps (no recurrence) ---
  k_gemm_bf16f32<<<dim3(G4H / 16, SLEN * BATCH / 32), 32, 0, stream>>>(
      xs_bf, eWih0b, gx0, G4H, EDIM);

  // --- encoder recurrence, software-pipelined across layers: 49 launches ---
  for (int t = 0; t <= SLEN; ++t) {
    k_enc_pipe<<<dim3(HDIM / 16, 2), 32, 0, stream>>>(
        t, gx0, e_b0, eWhh0b, hbuf0, c0, eWih1b, eWhh1b, e_b1, hbuf1, c1);
  }
  const bf16_t* h_enc0 = hbuf0 + (size_t)SLEN * BATCH * HDIM;
  const bf16_t* h_enc1 = hbuf1 + (size_t)SLEN * BATCH * HDIM;

  // --- decoder: h0 = tile(h_enc) -> compute h_enc@Whh^T once (32 rows) ---
  k_gemm_bf16f32<<<dim3(G4H / 16, 1), 32, 0, stream>>>(h_enc0, dWhh0b, hg0, G4H, HDIM);
  k_gemm_bf16f32<<<dim3(G4H / 16, 1), 32, 0, stream>>>(h_enc1, dWhh1b, hg1, G4H, HDIM);

  // layer 0: batched input GEMM + elementwise cell
  k_gemm_bf16f32<<<dim3(G4H / 16, DECM / 32), 32, 0, stream>>>(demb, dWih0b, gxd0, G4H, EDIM);
  k_dec_cell<<<zgrid(DECM * HDIM), ZT, 0, stream>>>(gxd0, hg0, dd_b0, c0, hd0b, nullptr, DECM);

  // layer 1
  k_gemm_bf16f32<<<dim3(G4H / 16, DECM / 32), 32, 0, stream>>>(hd0b, dWih1b, gxd1, G4H, HDIM);
  k_dec_cell<<<zgrid(DECM * HDIM), ZT, 0, stream>>>(gxd1, hg1, dd_b1, c1, nullptr, hd1f, DECM);

  // --- FC projection (memory-bound, fp32 WMMA, single pass over fc_W) ---
  k_fc_f32<<<dim3(VTGT / 16, DECM / 32), 32, 0, stream>>>(hd1f, fc_W, fc_b,
                                                          out + (size_t)BATCH * VTGT);
}
